// ResidualAttentionBlock_52802327937632
// MI455X (gfx1250) — matrix operand order repaired and checked by reading, not compiled, not run on hardware
//
#include <hip/hip_runtime.h>
#include <hip/hip_fp16.h>

typedef _Float16 f16;
typedef __attribute__((ext_vector_type(16))) _Float16 v16h;
typedef __attribute__((ext_vector_type(8)))  _Float16 v8h;
typedef __attribute__((ext_vector_type(4)))  _Float16 v4h;
typedef __attribute__((ext_vector_type(8)))  float    v8f;

#define D_MODEL 1024
#define N_HEAD  16
#define HEAD_DIM 64
#define SEQ     1024
#define BATCH   8
#define ROWS    (BATCH * SEQ)   /* 8192 */
#define NEG_BIG (-3.0e38f)

/* ---------------- async global->LDS (CDNA5 GLOBAL_LOAD_ASYNC_TO_LDS) -------- */
#if defined(__has_builtin)
# if __has_builtin(__builtin_amdgcn_global_load_async_to_lds_b128) && \
     __has_builtin(__builtin_amdgcn_s_wait_asynccnt)
#  define USE_ASYNC_LDS 1
# endif
#endif
#ifndef USE_ASYNC_LDS
# define USE_ASYNC_LDS 0
#endif

#if USE_ASYNC_LDS
typedef int v4i32 __attribute__((vector_size(16)));
typedef __attribute__((address_space(1))) v4i32 as1_v4i32;
typedef __attribute__((address_space(3))) v4i32 as3_v4i32;
/* copy 16 f16 (32B) global -> LDS, 16B-aligned both sides */
static __device__ __forceinline__ void copy16_g2l(f16* dst, const f16* src) {
  __builtin_amdgcn_global_load_async_to_lds_b128(
      (as1_v4i32*)(unsigned long long)src,
      (as3_v4i32*)(unsigned)(unsigned long long)dst, 0, 0);
  __builtin_amdgcn_global_load_async_to_lds_b128(
      (as1_v4i32*)((unsigned long long)src + 16),
      (as3_v4i32*)((unsigned)(unsigned long long)dst + 16), 0, 0);
}
static __device__ __forceinline__ void async_fence() {
  __builtin_amdgcn_s_wait_asynccnt(0);
}
#else
static __device__ __forceinline__ void copy16_g2l(f16* dst, const f16* src) {
  *(v8h*)dst = *(const v8h*)src;
  *(v8h*)(dst + 8) = *(const v8h*)(src + 8);
}
static __device__ __forceinline__ void async_fence() {}
#endif

/* ---------- WMMA fragment helpers (layouts per CDNA5 ISA 7.12.2) ---------- */
static __device__ __forceinline__ v16h pack16(const f16* lo, const f16* hi) {
  v8h a = *(const v8h*)lo;
  v8h b = *(const v8h*)hi;
  v16h r;
#pragma unroll
  for (int i = 0; i < 8; ++i) { r[i] = a[i]; r[8 + i] = b[i]; }
  return r;
}

/* A 16x32 f16: lanes 0-15 row=lane, K={0..7,16..23}; lanes 16-31 row=lane-16, K={8..15,24..31}. */
static __device__ __forceinline__ v16h load_a_frag(const f16* tile, int lda) {
  int lane = threadIdx.x & 31;
  int half = lane >> 4, r = lane & 15;
  const f16* p = tile + (size_t)r * lda + half * 8;
  return pack16(p, p + 16);
}

static __device__ __forceinline__ v8f wmma_f16(v16h a, v16h b, v8f c) {
  return __builtin_amdgcn_wmma_f32_16x16x32_f16(false, a, false, b, (short)0, c, false, false);
}

/* ---------- elementwise cast ---------- */
__global__ void cast_f32_to_f16_kernel(const float* __restrict__ in, f16* __restrict__ out, int n) {
  int i = blockIdx.x * blockDim.x + threadIdx.x;
  if (i < n) out[i] = (f16)in[i];
}

/* ---------- fused LayerNorm (fp32) + cast to f16; one block per row ---------- */
__global__ void __launch_bounds__(256)
ln_cast_kernel(const float* __restrict__ x, const float* __restrict__ g,
               const float* __restrict__ bta, f16* __restrict__ out) {
  int row = blockIdx.x;
  int t = threadIdx.x;
  const float4* xr = (const float4*)(x + (size_t)row * D_MODEL);
  float4 v = xr[t];
  float s  = v.x + v.y + v.z + v.w;
  float ss = v.x * v.x + v.y * v.y + v.z * v.z + v.w * v.w;
#pragma unroll
  for (int off = 1; off < 32; off <<= 1) {
    s  += __shfl_xor(s, off, 32);
    ss += __shfl_xor(ss, off, 32);
  }
  __shared__ float sbuf[8], ssbuf[8];
  int wv = t >> 5, ln = t & 31;
  if (ln == 0) { sbuf[wv] = s; ssbuf[wv] = ss; }
  __syncthreads();
  if (wv == 0) {
    float a  = (ln < 8) ? sbuf[ln]  : 0.f;
    float b2 = (ln < 8) ? ssbuf[ln] : 0.f;
#pragma unroll
    for (int off = 1; off < 8; off <<= 1) {
      a  += __shfl_xor(a, off, 32);
      b2 += __shfl_xor(b2, off, 32);
    }
    if (ln == 0) { sbuf[0] = a; ssbuf[0] = b2; }
  }
  __syncthreads();
  float mean = sbuf[0] * (1.0f / D_MODEL);
  float var  = ssbuf[0] * (1.0f / D_MODEL) - mean * mean;
  float inv  = rsqrtf(var + 1e-5f);
  int c = t * 4;
  v4h o;
  o[0] = (f16)((v.x - mean) * inv * g[c + 0] + bta[c + 0]);
  o[1] = (f16)((v.y - mean) * inv * g[c + 1] + bta[c + 1]);
  o[2] = (f16)((v.z - mean) * inv * g[c + 2] + bta[c + 2]);
  o[3] = (f16)((v.w - mean) * inv * g[c + 3] + bta[c + 3]);
  *(v4h*)(out + (size_t)row * D_MODEL + c) = o;
}

/* ---------- WMMA GEMM: C[M,N] = A[M,K] * W[N,K]^T + bias ----------
   Block 256 threads = 8 waves; block tile 128x128; wave tile 32x64
   (waves: mw = wave>>1 in 0..3 over M, nw = wave&1 over N).
   K staged in double-buffered LDS (async on CDNA5), BK = 32.
   EPI 0: store f16 | EPI 1: +resid, store f32 | EPI 2: quickGELU, f16. */
#define BM 128
#define BN 128
#define BK 32
#define SPAD 40   /* padded LDS row stride (f16 elems): 80B spreads banks */

template <int EPI>
__global__ void __launch_bounds__(256)
gemm_f16_kernel(const f16* __restrict__ A, int lda,
                const f16* __restrict__ W, int ldw,
                const float* __restrict__ bias,
                const float* __restrict__ resid,
                float* __restrict__ outF, f16* __restrict__ outH,
                int M, int N, int K) {
  __shared__ f16 As[2][BM * SPAD];
  __shared__ f16 Bs[2][BN * SPAD];

  int tid  = threadIdx.x;
  int wave = tid >> 5;
  int lane = tid & 31;
  int half = lane >> 4, ln = lane & 15;
  int mw = wave >> 1;   /* 0..3 */
  int nw = wave & 1;    /* 0..1 */

  int n0 = blockIdx.x * BN;
  int m0 = blockIdx.y * BM;

  /* staging role: each of 256 threads moves 16 f16 of A and 16 of B per k-step */
  int srow  = tid >> 1;   /* 0..127 */
  int shalf = tid & 1;    /* 0..1   */
  const f16* gA = A + (size_t)(m0 + srow) * lda + shalf * 16;
  const f16* gW = W + (size_t)(n0 + srow) * ldw + shalf * 16;
  f16* sA = &As[0][srow * SPAD + shalf * 16];
  f16* sB = &Bs[0][srow * SPAD + shalf * 16];
  const int sbufstride_a = BM * SPAD;
  const int sbufstride_b = BN * SPAD;

  v8f acc[2][4];
#pragma unroll
  for (int mt = 0; mt < 2; ++mt)
#pragma unroll
    for (int nt = 0; nt < 4; ++nt)
#pragma unroll
      for (int r = 0; r < 8; ++r) acc[mt][nt][r] = 0.f;

  /* prologue: fill buffer 0 */
  copy16_g2l(sA, gA);
  copy16_g2l(sB, gW);
  async_fence();
  __syncthreads();

  for (int k0 = 0; k0 < K; k0 += BK) {
    int cur = (k0 / BK) & 1;
    /* prefetch next k-tile into the other buffer */
    if (k0 + BK < K) {
      copy16_g2l(sA + (cur ^ 1) * sbufstride_a, gA + (k0 + BK));
      copy16_g2l(sB + (cur ^ 1) * sbufstride_b, gW + (k0 + BK));
    }

    const f16* as = As[cur];
    const f16* bs = Bs[cur];
    v16h af[2], bf[4];
#pragma unroll
    for (int mt = 0; mt < 2; ++mt) {
      const f16* p = as + (mw * 32 + mt * 16 + ln) * SPAD + half * 8;
      af[mt] = pack16(p, p + 16);
    }
#pragma unroll
    for (int nt = 0; nt < 4; ++nt) {
      const f16* p = bs + (nw * 64 + nt * 16 + ln) * SPAD + half * 8;
      bf[nt] = pack16(p, p + 16);
    }
#pragma unroll
    for (int mt = 0; mt < 2; ++mt)
#pragma unroll
      for (int nt = 0; nt < 4; ++nt)
        acc[mt][nt] = wmma_f16(af[mt], bf[nt], acc[mt][nt]);

    async_fence();
    __syncthreads();
  }

  /* epilogue */
#pragma unroll
  for (int mt = 0; mt < 2; ++mt)
#pragma unroll
    for (int nt = 0; nt < 4; ++nt) {
      int col = n0 + nw * 64 + nt * 16 + ln;
      float bv = bias[col];
#pragma unroll
      for (int r = 0; r < 8; ++r) {
        int row = m0 + mw * 32 + mt * 16 + half * 8 + r;
        size_t idx = (size_t)row * N + col;
        float v = acc[mt][nt][r] + bv;
        if (EPI == 0) {
          outH[idx] = (f16)v;
        } else if (EPI == 1) {
          outF[idx] = v + resid[idx];
        } else {
          float gg = v * (1.0f / (1.0f + __expf(-1.702f * v)));
          outH[idx] = (f16)gg;
        }
      }
    }
}

/* ---------- flash attention: softmax(QK^T/8)V ----------
   grid = B*H*(SEQ/64); block = 128 (4 waves, 16 q-rows each). */
__global__ void __launch_bounds__(128)
attn_kernel(const f16* __restrict__ qkv, f16* __restrict__ ctx) {
  __shared__ f16 ldsK[32 * 64];       /* K chunk  [key][hd]   */
  __shared__ f16 ldsVT[64 * 32];      /* V chunk  [hd][key]   */
  __shared__ f16 ldsP[4][16 * 32];    /* per-wave P tile [row][key] */

  int bh = blockIdx.x / (SEQ / 64);
  int qt = blockIdx.x % (SEQ / 64);
  int b = bh / N_HEAD, h = bh % N_HEAD;
  int wave = threadIdx.x >> 5, lane = threadIdx.x & 31;
  int half = lane >> 4, ln = lane & 15;
  int m0 = qt * 64 + wave * 16;

  const int ldq = 3 * D_MODEL;
  const f16* qbase = qkv + (size_t)(b * SEQ + m0) * ldq + h * HEAD_DIM;
  v16h qa0 = load_a_frag(qbase, ldq);        /* hd 0..31  */
  v16h qa1 = load_a_frag(qbase + 32, ldq);   /* hd 32..63 */

  v8f acc[4];
  float mrun[8], lrun[8];
#pragma unroll
  for (int nt = 0; nt < 4; ++nt)
#pragma unroll
    for (int r = 0; r < 8; ++r) acc[nt][r] = 0.f;
#pragma unroll
  for (int r = 0; r < 8; ++r) { mrun[r] = NEG_BIG; lrun[r] = 0.f; }

  for (int kc = 0; kc < SEQ; kc += 32) {
    /* --- stage K (async, row major) and V (transposed scatter) --- */
    {
      int t = threadIdx.x;
      int key = t >> 2;
      int hd = (t & 3) * 16;
      const f16* gk = qkv + (size_t)(b * SEQ + kc + key) * ldq + D_MODEL + h * HEAD_DIM + hd;
      copy16_g2l(&ldsK[key * 64 + hd], gk);
      const f16* gv = gk + D_MODEL;
      v8h vlo = *(const v8h*)gv;
      v8h vhi = *(const v8h*)(gv + 8);
#pragma unroll
      for (int j = 0; j < 8; ++j) {
        ldsVT[(hd + j) * 32 + key]     = vlo[j];
        ldsVT[(hd + 8 + j) * 32 + key] = vhi[j];
      }
    }
    async_fence();
    __syncthreads();

    /* --- scores: two 16x16 key tiles, contraction over hd --- */
    v8f s[2];
#pragma unroll
    for (int tile = 0; tile < 2; ++tile) {
      v8f z;
#pragma unroll
      for (int r = 0; r < 8; ++r) z[r] = 0.f;
      const f16* kp = &ldsK[(tile * 16 + ln) * 64 + half * 8];
      v16h b0 = pack16(kp, kp + 16);           /* hd 0..31  */
      v16h b1 = pack16(kp + 32, kp + 48);      /* hd 32..63 */
      z = wmma_f16(qa0, b0, z);
      z = wmma_f16(qa1, b1, z);
      s[tile] = z;
    }

    /* --- online softmax over the 32-key chunk --- */
    float mx[8];
#pragma unroll
    for (int r = 0; r < 8; ++r) {
      s[0][r] *= 0.125f;
      s[1][r] *= 0.125f;
      mx[r] = fmaxf(s[0][r], s[1][r]);
    }
#pragma unroll
    for (int off = 1; off < 16; off <<= 1)
#pragma unroll
      for (int r = 0; r < 8; ++r) mx[r] = fmaxf(mx[r], __shfl_xor(mx[r], off, 32));

    float p0[8], p1[8], rs[8];
#pragma unroll
    for (int r = 0; r < 8; ++r) {
      float mn = fmaxf(mrun[r], mx[r]);
      float alpha = __expf(mrun[r] - mn);
      mrun[r] = mn;
      lrun[r] *= alpha;
      acc[0][r] *= alpha; acc[1][r] *= alpha; acc[2][r] *= alpha; acc[3][r] *= alpha;
      p0[r] = __expf(s[0][r] - mn);
      p1[r] = __expf(s[1][r] - mn);
      rs[r] = p0[r] + p1[r];
    }
#pragma unroll
    for (int off = 1; off < 16; off <<= 1)
#pragma unroll
      for (int r = 0; r < 8; ++r) rs[r] += __shfl_xor(rs[r], off, 32);
#pragma unroll
    for (int r = 0; r < 8; ++r) lrun[r] += rs[r];

    /* --- C-layout -> A-layout for P via per-wave LDS round trip --- */
    f16* pb = ldsP[wave];
#pragma unroll
    for (int r = 0; r < 8; ++r) {
      pb[(half * 8 + r) * 32 + ln]      = (f16)p0[r];
      pb[(half * 8 + r) * 32 + 16 + ln] = (f16)p1[r];
    }
    __syncthreads();

    /* --- ctx += P (16x32) @ V (32x64) --- */
    {
      const f16* pp = ldsP[wave] + ln * 32 + half * 8;
      v16h pa = pack16(pp, pp + 16);
#pragma unroll
      for (int nt = 0; nt < 4; ++nt) {
        const f16* vp = &ldsVT[(nt * 16 + ln) * 32 + half * 8];
        v16h vb = pack16(vp, vp + 16);
        acc[nt] = wmma_f16(pa, vb, acc[nt]);
      }
    }
    __syncthreads();
  }

  /* --- normalize and write ctx --- */
#pragma unroll
  for (int nt = 0; nt < 4; ++nt)
#pragma unroll
    for (int r = 0; r < 8; ++r) {
      int row = m0 + half * 8 + r;
      ctx[(size_t)(b * SEQ + row) * D_MODEL + h * HEAD_DIM + nt * 16 + ln] =
          (f16)(acc[nt][r] / lrun[r]);
    }
}

/* ---------- host side ---------- */
extern "C" void kernel_launch(void* const* d_in, const int* in_sizes, int n_in,
                              void* d_out, int out_size, void* d_ws, size_t ws_size,
                              hipStream_t stream) {
  const float* x         = (const float*)d_in[0];
  const float* in_proj_w = (const float*)d_in[1];
  const float* in_proj_b = (const float*)d_in[2];
  const float* out_w     = (const float*)d_in[3];
  const float* out_b     = (const float*)d_in[4];
  const float* ln1_g     = (const float*)d_in[5];
  const float* ln1_b     = (const float*)d_in[6];
  const float* ln2_g     = (const float*)d_in[7];
  const float* ln2_b     = (const float*)d_in[8];
  const float* fc_w      = (const float*)d_in[9];
  const float* fc_b      = (const float*)d_in[10];
  const float* proj_w    = (const float*)d_in[11];
  const float* proj_b    = (const float*)d_in[12];
  float* out = (float*)d_out;

  const int D = D_MODEL;
  char* ws = (char*)d_ws;
  size_t off = 0;
  auto alloc = [&](size_t bytes) -> char* {
    char* p = ws + off;
    off += (bytes + 255) & ~(size_t)255;
    return p;
  };

  f16* h16   = (f16*)alloc((size_t)ROWS * D * 2);       /* LN1 out, reused for LN2 out */
  f16* big   = (f16*)alloc((size_t)ROWS * 4 * D * 2);   /* qkv16 (48MB) then hid16 (64MB) */
  f16* qkv16 = big;                                     /* [8192, 3072] */
  f16* hid16 = big;                                     /* [8192, 4096], after qkv dead  */
  f16* ctx16 = (f16*)alloc((size_t)ROWS * D * 2);
  float* x2  = (float*)alloc((size_t)ROWS * D * 4);
  f16* wqkv  = (f16*)alloc((size_t)3 * D * D * 2);
  f16* wout  = (f16*)alloc((size_t)D * D * 2);
  f16* wfc   = (f16*)alloc((size_t)4 * D * D * 2);
  f16* wproj = (f16*)alloc((size_t)4 * D * D * 2);

  /* 1) cast weights fp32 -> f16 */
  {
    int n;
    n = 3 * D * D; cast_f32_to_f16_kernel<<<(n + 255) / 256, 256, 0, stream>>>(in_proj_w, wqkv, n);
    n = D * D;     cast_f32_to_f16_kernel<<<(n + 255) / 256, 256, 0, stream>>>(out_w, wout, n);
    n = 4 * D * D; cast_f32_to_f16_kernel<<<(n + 255) / 256, 256, 0, stream>>>(fc_w, wfc, n);
    n = 4 * D * D; cast_f32_to_f16_kernel<<<(n + 255) / 256, 256, 0, stream>>>(proj_w, wproj, n);
  }

  /* 2) LN1 */
  ln_cast_kernel<<<ROWS, 256, 0, stream>>>(x, ln1_g, ln1_b, h16);

  /* 3) QKV projection: [8192,1024] x [3072,1024]^T -> f16 */
  gemm_f16_kernel<0><<<dim3(3 * D / BN, ROWS / BM), 256, 0, stream>>>(
      h16, D, wqkv, D, in_proj_b, nullptr, nullptr, qkv16, ROWS, 3 * D, D);

  /* 4) attention */
  attn_kernel<<<BATCH * N_HEAD * (SEQ / 64), 128, 0, stream>>>(qkv16, ctx16);

  /* 5) out projection + residual -> x2 (f32) */
  gemm_f16_kernel<1><<<dim3(D / BN, ROWS / BM), 256, 0, stream>>>(
      ctx16, D, wout, D, out_b, x, x2, nullptr, ROWS, D, D);

  /* 6) LN2 */
  ln_cast_kernel<<<ROWS, 256, 0, stream>>>(x2, ln2_g, ln2_b, h16);

  /* 7) FC + quickGELU -> hid16 (f16) */
  gemm_f16_kernel<2><<<dim3(4 * D / BN, ROWS / BM), 256, 0, stream>>>(
      h16, D, wfc, D, fc_b, nullptr, nullptr, hid16, ROWS, 4 * D, D);

  /* 8) MLP projection + residual -> d_out (f32) */
  gemm_f16_kernel<1><<<dim3(D / BN, ROWS / BM), 256, 0, stream>>>(
      hid16, 4 * D, wproj, 4 * D, proj_b, x2, out, nullptr, ROWS, D, 4 * D);
}
